// GroupedQueryAttention_9079560864167
// MI455X (gfx1250) — matrix operand order repaired and checked by reading, not compiled, not run on hardware
//
#include <hip/hip_runtime.h>

// ---------------------------------------------------------------------------
// Types for CDNA5 WMMA fragments (wave32)
// ---------------------------------------------------------------------------
typedef __attribute__((ext_vector_type(16))) _Float16 v16h;
typedef __attribute__((ext_vector_type(8)))  _Float16 v8h;
typedef __attribute__((ext_vector_type(8)))  float    v8f;

#define S_LEN 2048
#define D_DIM 2048
#define N_H   32
#define N_G   8
#define HDIM  64
#define KV_D  (N_G * HDIM)   // 512

__device__ __forceinline__ v16h cat8(v8h lo, v8h hi) {
  return __builtin_shufflevector(lo, hi, 0, 1, 2, 3, 4, 5, 6, 7,
                                         8, 9, 10, 11, 12, 13, 14, 15);
}

__device__ __forceinline__ v8f wmma_f16(v16h a, v16h b, v8f c) {
  // D = A(16x32 f16) * B(32x16 f16) + C(16x16 f32)
  return __builtin_amdgcn_wmma_f32_16x16x32_f16(
      /*neg_a=*/false, a, /*neg_b=*/false, b,
      /*c_mod=*/(short)0, c, /*reuse_a=*/false, /*reuse_b=*/false);
}

// ---------------------------------------------------------------------------
// f32 -> f16 cast
// ---------------------------------------------------------------------------
__global__ void cast_f32_to_f16(const float* __restrict__ in,
                                _Float16* __restrict__ out, int n) {
  int i = blockIdx.x * blockDim.x + threadIdx.x;
  if (i < n) out[i] = (_Float16)in[i];
}

// ---------------------------------------------------------------------------
// RoPE (rotate-half), in place on [S, nheads, 64] f16.  cos/sin are [S, 64] f32.
// ---------------------------------------------------------------------------
__global__ void rope_kernel(_Float16* __restrict__ t,
                            const float* __restrict__ cosv,
                            const float* __restrict__ sinv, int nheads) {
  int idx = blockIdx.x * blockDim.x + threadIdx.x;   // over S*nheads*32 pairs
  int d = idx & 31;
  int h = (idx >> 5) % nheads;
  int s = idx / (32 * nheads);
  if (s >= S_LEN) return;
  float c1 = cosv[s * HDIM + d];
  float s1 = sinv[s * HDIM + d];
  float c2 = cosv[s * HDIM + d + 32];
  float s2 = sinv[s * HDIM + d + 32];
  _Float16* row = t + ((size_t)s * nheads + h) * HDIM;
  float t1 = (float)row[d];
  float t2 = (float)row[d + 32];
  row[d]      = (_Float16)(t1 * c1 - t2 * s1);
  row[d + 32] = (_Float16)(t2 * c2 + t1 * s2);
}

// ---------------------------------------------------------------------------
// V [S, G*64] -> Vt [G][64][S]   (so PV B-fragments are contiguous along keys)
// ---------------------------------------------------------------------------
__global__ void transpose_v_kernel(const _Float16* __restrict__ v,
                                   _Float16* __restrict__ vt) {
  int idx = blockIdx.x * blockDim.x + threadIdx.x;   // over S*KV_D
  if (idx >= S_LEN * KV_D) return;
  int d = idx & (HDIM - 1);
  int g = (idx / HDIM) & (N_G - 1);
  int s = idx / KV_D;
  vt[((size_t)(g * HDIM + d)) * S_LEN + s] = v[(size_t)s * KV_D + g * HDIM + d];
}

// ---------------------------------------------------------------------------
// GEMM: C[M,N] = A[M,Kd](row-major f16) * Bw[N,Kd](row-major f16)^T
//
// Register-blocked: each wave owns a 32x64 tile = 2x4 WMMA fragments.
// Per 32-wide k-step: 6 fragment loads (6 KB) feed 8 WMMAs (128 KFLOP)
// -> ~21 FLOP/byte register-level intensity, 8 independent accumulation
// chains to hide XDL latency.  4 waves/block tile 64x128.
//
// A-fragment: lane L holds row (L%16), K-halves at (L<16?0:8)+{0..7} and
//             16+(L<16?0:8)+{0..7}  -> two 16B loads.
// B-fragment: lane L holds column (L%16), K-halves at (L<16?0:8)+{0..7} and
//             16+(L<16?0:8)+{0..7} -> two 16B loads straight from the weight row.
// ---------------------------------------------------------------------------
__global__ __launch_bounds__(128) void gemm_wmma_kernel(
    const _Float16* __restrict__ A, const _Float16* __restrict__ Bw,
    _Float16* __restrict__ Ch, float* __restrict__ Cf, int Ndim, int Kd) {
  const int lane = threadIdx.x & 31;
  const int wave = threadIdx.x >> 5;
  const int lm = lane & 15;
  const int hi = lane >> 4;
  const int m0 = blockIdx.y * 64 + (wave >> 1) * 32;
  const int n0 = blockIdx.x * 128 + (wave & 1) * 64;

  const _Float16* pa = A + (size_t)(m0 + lm) * Kd + hi * 8;
  const _Float16* pb = Bw + (size_t)(n0 + lm) * Kd + hi * 8;

  v8f acc[2][4] = {};
  for (int k0 = 0; k0 < Kd; k0 += 32) {
    v16h av[2];
    v16h bv[4];
#pragma unroll
    for (int i = 0; i < 2; ++i) {
      const _Float16* p = pa + (size_t)(16 * i) * Kd + k0;
      av[i] = cat8(*(const v8h*)p, *(const v8h*)(p + 16));
    }
#pragma unroll
    for (int j = 0; j < 4; ++j) {
      const _Float16* p = pb + (size_t)(16 * j) * Kd + k0;
      bv[j] = cat8(*(const v8h*)p, *(const v8h*)(p + 16));
    }
    __builtin_prefetch(pa + k0 + 512, 0, 1);
    __builtin_prefetch(pb + k0 + 512, 0, 1);
#pragma unroll
    for (int i = 0; i < 2; ++i)
#pragma unroll
      for (int j = 0; j < 4; ++j)
        acc[i][j] = wmma_f16(av[i], bv[j], acc[i][j]);
  }

#pragma unroll
  for (int i = 0; i < 2; ++i)
#pragma unroll
    for (int j = 0; j < 4; ++j)
#pragma unroll
      for (int r = 0; r < 8; ++r) {
        const size_t idx =
            (size_t)(m0 + 16 * i + r + hi * 8) * Ndim + n0 + 16 * j + lm;
        if (Ch != nullptr)
          Ch[idx] = (_Float16)acc[i][j][r];
        else
          Cf[idx] = acc[i][j][r];
      }
}

// ---------------------------------------------------------------------------
// Flash attention, one wave per (head, 16-query tile).
//   q   : [S, H*64]  f16 (RoPE applied)
//   k   : [S, G*64]  f16 (RoPE applied)
//   vt  : [G][64][S] f16
//   att : [S, H*64]  f16
// Scores/softmax in f32 C-fragments; online softmax with shfl_xor row
// reductions (rows of a 16x16 C tile live across 16-lane halves).
// P is re-striped C-layout -> A-layout through per-wave LDS.
// ---------------------------------------------------------------------------
__global__ __launch_bounds__(128) void attn_kernel(
    const _Float16* __restrict__ q, const _Float16* __restrict__ k,
    const _Float16* __restrict__ vt, _Float16* __restrict__ att) {
  __shared__ _Float16 lds[4 * 16 * 32];

  const int lane = threadIdx.x & 31;
  const int wave = threadIdx.x >> 5;
  const int lm = lane & 15;
  const int hi = lane >> 4;
  const int h = blockIdx.x;       // head
  const int g = h >> 2;           // GS = H/G = 4
  const int q0 = (blockIdx.y * 4 + wave) * 16;

  // Q fragments: A-layout, K = head-dim, two 32-wide chunks.
  v16h qa[2];
  const _Float16* qrow = q + (size_t)(q0 + lm) * (N_H * HDIM) + h * HDIM;
#pragma unroll
  for (int c = 0; c < 2; ++c) {
    const _Float16* p = qrow + c * 32 + hi * 8;
    qa[c] = cat8(*(const v8h*)p, *(const v8h*)(p + 16));
  }

  float mrow[8], lrow[8];
  v8f o[4] = {};
#pragma unroll
  for (int r = 0; r < 8; ++r) {
    mrow[r] = -3.0e38f;
    lrow[r] = 0.0f;
  }

  _Float16* pl = lds + wave * 512;

  for (int kt0 = 0; kt0 < q0 + 16; kt0 += 32) {
    // ---- scores for 32 keys: two 16x16 WMMA tiles -------------------------
    v8f sc[2];
#pragma unroll
    for (int t = 0; t < 2; ++t) {
      const int n0 = kt0 + t * 16;
      v8f a = {};
      const _Float16* krow = k + (size_t)(n0 + lm) * KV_D + g * HDIM + hi * 8;
#pragma unroll
      for (int c = 0; c < 2; ++c) {
        v8h b0 = *(const v8h*)(krow + c * 32);
        v8h b1 = *(const v8h*)(krow + c * 32 + 16);
        a = wmma_f16(qa[c], cat8(b0, b1), a);
      }
      sc[t] = a;
    }

    // ---- scale, causal mask, online softmax -------------------------------
#pragma unroll
    for (int r = 0; r < 8; ++r) {
      const int m = q0 + r + hi * 8;
      float a = sc[0][r] * 0.125f;               // 1/sqrt(64)
      if (kt0 + lm > m) a = -3.0e38f;
      float b = sc[1][r] * 0.125f;
      if (kt0 + 16 + lm > m) b = -3.0e38f;

      float tmax = fmaxf(a, b);
#pragma unroll
      for (int msk = 1; msk < 16; msk <<= 1)
        tmax = fmaxf(tmax, __shfl_xor(tmax, msk, 32));
      const float newm = fmaxf(mrow[r], tmax);

      const float p0 = __expf(a - newm);
      const float p1 = __expf(b - newm);
      float s = p0 + p1;
#pragma unroll
      for (int msk = 1; msk < 16; msk <<= 1) s += __shfl_xor(s, msk, 32);

      const float resc = __expf(mrow[r] - newm);
      lrow[r] = lrow[r] * resc + s;
      mrow[r] = newm;
#pragma unroll
      for (int f = 0; f < 4; ++f) o[f][r] *= resc;

      // C-layout -> row-major [16 rows][32 keys] P tile in LDS (f16)
      pl[(r + hi * 8) * 32 + lm] = (_Float16)p0;
      pl[(r + hi * 8) * 32 + 16 + lm] = (_Float16)p1;
    }
    asm volatile("s_wait_dscnt 0" ::: "memory");  // LDS RAW within the wave

    // ---- reload P as an A-fragment (16x32, K = keys) ----------------------
    const _Float16* pr = pl + lm * 32 + hi * 8;
    v16h pv = cat8(*(const v8h*)pr, *(const v8h*)(pr + 16));

    // ---- O += P * V  (4 fragments cover head_dim 64) ----------------------
#pragma unroll
    for (int f = 0; f < 4; ++f) {
      const _Float16* vrow =
          vt + (size_t)(g * HDIM + f * 16 + lm) * S_LEN + kt0 + hi * 8;
      v8h b0 = *(const v8h*)vrow;
      v8h b1 = *(const v8h*)(vrow + 16);
      o[f] = wmma_f16(pv, cat8(b0, b1), o[f]);
    }
  }

  // ---- normalize and write att[q, h*64 + d] in f16 ------------------------
#pragma unroll
  for (int f = 0; f < 4; ++f)
#pragma unroll
    for (int r = 0; r < 8; ++r)
      att[(size_t)(q0 + r + hi * 8) * (N_H * HDIM) + h * HDIM + f * 16 + lm] =
          (_Float16)(o[f][r] / lrow[r]);
}

// ---------------------------------------------------------------------------
// Launcher
// ---------------------------------------------------------------------------
extern "C" void kernel_launch(void* const* d_in, const int* in_sizes, int n_in,
                              void* d_out, int out_size, void* d_ws,
                              size_t ws_size, hipStream_t stream) {
  const float* x = (const float*)d_in[0];
  // d_in[1] = mask (bool) -- causal mask is computed analytically, unused
  const float* cosv = (const float*)d_in[2];
  const float* sinv = (const float*)d_in[3];
  const float* Wq = (const float*)d_in[4];
  const float* Wk = (const float*)d_in[5];
  const float* Wv = (const float*)d_in[6];
  const float* Wo = (const float*)d_in[7];
  float* out = (float*)d_out;

  _Float16* ws = (_Float16*)d_ws;
  size_t off = 0;
  _Float16* xh = ws + off;  off += (size_t)S_LEN * D_DIM;
  _Float16* Wqh = ws + off; off += (size_t)D_DIM * D_DIM;
  _Float16* Wkh = ws + off; off += (size_t)KV_D * D_DIM;
  _Float16* Wvh = ws + off; off += (size_t)KV_D * D_DIM;
  _Float16* Woh = ws + off; off += (size_t)D_DIM * D_DIM;
  _Float16* qb = ws + off;  off += (size_t)S_LEN * D_DIM;
  _Float16* kb = ws + off;  off += (size_t)S_LEN * KV_D;
  _Float16* vb = ws + off;  off += (size_t)S_LEN * KV_D;
  _Float16* vtb = ws + off; off += (size_t)S_LEN * KV_D;
  _Float16* attb = ws + off;

  const int TB = 256;
  auto grid1 = [&](int n) { return dim3((n + TB - 1) / TB); };

  // 1) casts to f16
  cast_f32_to_f16<<<grid1(S_LEN * D_DIM), TB, 0, stream>>>(x, xh, S_LEN * D_DIM);
  cast_f32_to_f16<<<grid1(D_DIM * D_DIM), TB, 0, stream>>>(Wq, Wqh, D_DIM * D_DIM);
  cast_f32_to_f16<<<grid1(KV_D * D_DIM), TB, 0, stream>>>(Wk, Wkh, KV_D * D_DIM);
  cast_f32_to_f16<<<grid1(KV_D * D_DIM), TB, 0, stream>>>(Wv, Wvh, KV_D * D_DIM);
  cast_f32_to_f16<<<grid1(D_DIM * D_DIM), TB, 0, stream>>>(Wo, Woh, D_DIM * D_DIM);

  // 2) QKV projections (WMMA, 64x128 block tiles / 32x64 wave tiles)
  gemm_wmma_kernel<<<dim3(D_DIM / 128, S_LEN / 64), 128, 0, stream>>>(
      xh, Wqh, qb, nullptr, D_DIM, D_DIM);
  gemm_wmma_kernel<<<dim3(KV_D / 128, S_LEN / 64), 128, 0, stream>>>(
      xh, Wkh, kb, nullptr, KV_D, D_DIM);
  gemm_wmma_kernel<<<dim3(KV_D / 128, S_LEN / 64), 128, 0, stream>>>(
      xh, Wvh, vb, nullptr, KV_D, D_DIM);

  // 3) RoPE on q and k
  rope_kernel<<<grid1(S_LEN * N_H * 32), TB, 0, stream>>>(qb, cosv, sinv, N_H);
  rope_kernel<<<grid1(S_LEN * N_G * 32), TB, 0, stream>>>(kb, cosv, sinv, N_G);

  // 4) V transpose for PV fragment layout
  transpose_v_kernel<<<grid1(S_LEN * KV_D), TB, 0, stream>>>(vb, vtb);

  // 5) flash attention (WMMA QK^T and PV, online softmax)
  attn_kernel<<<dim3(N_H, S_LEN / 64), 128, 0, stream>>>(qb, kb, vtb, attb);

  // 6) output projection, f32 result
  gemm_wmma_kernel<<<dim3(D_DIM / 128, S_LEN / 64), 128, 0, stream>>>(
      attb, Woh, nullptr, out, D_DIM, D_DIM);
}
